// Capsule_5403068858451
// MI455X (gfx1250) — matrix operand order repaired and checked by reading, not compiled, not run on hardware
//
#include <hip/hip_runtime.h>
#include <hip/hip_bf16.h>

#define BQ 32
#define NQ 4096
#define DQ 256
#define IC 32      // NUM_CAPSULE
#define JC 16      // DIM_CAPSULE
#define CHUNKS 8   // routing n-chunks per batch

typedef __attribute__((ext_vector_type(16))) __bf16 v16bf;
typedef __attribute__((ext_vector_type(8)))  float  v8f;

// ---------------------------------------------------------------------------
// K0: pack W [D=256, E=512] f32 row-major into WMMA B-fragment order (bf16).
// Layout: Wp[((nt*8 + kb)*32 + lane)*16 + q]
//   lane<16 : column n = nt*16+lane,   k = kb*32 + {0..7, 16..23}   (q=0..15)
//   lane>=16: column n = nt*16+lane-16, k = kb*32 + {8..15, 24..31}
// ---------------------------------------------------------------------------
__global__ __launch_bounds__(256) void pack_w(const float* __restrict__ W,
                                              __bf16* __restrict__ Wp) {
    int t = blockIdx.x * 256 + threadIdx.x;     // 0..131071
    int q    = t & 15;
    int lane = (t >> 4) & 31;
    int kb   = (t >> 9) & 7;
    int nt   = t >> 12;                         // 0..31
    int n = nt * 16 + (lane & 15);
    int k = kb * 32 + ((lane >> 4) ? 8 : 0) + (q < 8 ? q : q + 8);
    Wp[t] = (__bf16)W[k * (IC * JC) + n];
}

// ---------------------------------------------------------------------------
// K1: u_hat = u_vecs @ W, stored bf16 transposed as [B, I, N, J].
// 4 waves / block share one 16-row A tile; wave w owns capsules w*8..w*8+7.
// ---------------------------------------------------------------------------
__global__ __launch_bounds__(128) void gemm_uhat(const float* __restrict__ u_vecs,
                                                 const __bf16* __restrict__ Wp,
                                                 __bf16* __restrict__ uhat) {
    const int mtile = blockIdx.x;               // 0..8191
    const int wave  = threadIdx.x >> 5;
    const int lane  = threadIdx.x & 31;
    const int row0  = mtile * 16;
    const int m     = lane & 15;
    const int khalf = lane >> 4;
    const float* arow = u_vecs + (long)(row0 + m) * DQ;

    v8f acc[8] = {};

#pragma unroll
    for (int kb = 0; kb < 8; ++kb) {
        // A fragment: lane<16 holds K = kb*32 + {0..7, 16..23}
        //             lane>=16 holds K = kb*32 + {8..15, 24..31}
        const int k0 = kb * 32 + khalf * 8;
        float4 a0 = *(const float4*)(arow + k0);
        float4 a1 = *(const float4*)(arow + k0 + 4);
        float4 a2 = *(const float4*)(arow + k0 + 16);
        float4 a3 = *(const float4*)(arow + k0 + 20);
        v16bf af;
        af[0]=(__bf16)a0.x;  af[1]=(__bf16)a0.y;  af[2]=(__bf16)a0.z;  af[3]=(__bf16)a0.w;
        af[4]=(__bf16)a1.x;  af[5]=(__bf16)a1.y;  af[6]=(__bf16)a1.z;  af[7]=(__bf16)a1.w;
        af[8]=(__bf16)a2.x;  af[9]=(__bf16)a2.y;  af[10]=(__bf16)a2.z; af[11]=(__bf16)a2.w;
        af[12]=(__bf16)a3.x; af[13]=(__bf16)a3.y; af[14]=(__bf16)a3.z; af[15]=(__bf16)a3.w;

#pragma unroll
        for (int t = 0; t < 8; ++t) {
            const int nt = wave * 8 + t;
            const v16bf bf = *(const v16bf*)(Wp + (((nt * 8 + kb) * 32 + lane) << 4));
            acc[t] = __builtin_amdgcn_wmma_f32_16x16x32_bf16(
                false, af, false, bf, (short)0, acc[t], false, false);
        }
    }

    // Scatter C tiles into u_hat[b][i][n][j] (bf16).
    const int b  = row0 / NQ;                   // 4096 % 16 == 0: uniform per block
    const int n0 = row0 % NQ;
    const int j  = lane & 15;
    const int radd = (lane >> 4) * 8;
#pragma unroll
    for (int t = 0; t < 8; ++t) {
        const int i = wave * 8 + t;
        __bf16* ob = uhat + (((long)(b * IC + i) * NQ + n0) * JC);
#pragma unroll
        for (int r = 0; r < 8; ++r)
            ob[(long)(r + radd) * JC + j] = (__bf16)acc[t][r];
    }
}

// ---------------------------------------------------------------------------
// K2: routing iter 0 — b==0 => c = 1/32; o = mean_n(u_hat)/1 * (1/32) ... then
// L2-normalize over j. One block per (b,i).
// ---------------------------------------------------------------------------
__global__ __launch_bounds__(256) void o_init(const __bf16* __restrict__ uhat,
                                              float* __restrict__ o_norm) {
    const int bi = blockIdx.x;                  // b*32+i
    const int j  = threadIdx.x & 15;
    const int g  = threadIdx.x >> 4;            // 16 n-groups
    const __bf16* base = uhat + (long)bi * NQ * JC;
    float s = 0.f;
    for (int n = g; n < NQ; n += 16) s += (float)base[n * JC + j];
    __shared__ float red[16][16];
    red[g][j] = s;
    __syncthreads();
    if (g == 0) {
        float tot = 0.f;
#pragma unroll
        for (int gg = 0; gg < 16; ++gg) tot += red[gg][j];
        tot *= (1.0f / 32.0f);
        float s2 = tot * tot;
#pragma unroll
        for (int msk = 1; msk < 16; msk <<= 1) s2 += __shfl_xor(s2, msk, 16);
        o_norm[bi * JC + j] = tot * rsqrtf(fmaxf(s2, 1e-12f));
    }
}

__global__ void zero_o(float* __restrict__ p) {
    p[blockIdx.x * 256 + threadIdx.x] = 0.f;
}

// ---------------------------------------------------------------------------
// K4: one fused routing iteration.
//   b[i,n] = <o_prev[i,:], u_hat[i,n,:]> ; c = softmax_i(b) ;
//   o_next[i,:] += sum_n c[i,n]*u_hat[i,n,:]
// Lane layout: j = tid&15, 16-lane group owns one n per step.
// ---------------------------------------------------------------------------
__global__ __launch_bounds__(256) void route_step(const __bf16* __restrict__ uhat,
                                                  const float* __restrict__ o_prev,
                                                  float* __restrict__ o_next) {
    const int b     = blockIdx.x;               // 0..31
    const int chunk = blockIdx.y;               // 0..CHUNKS-1
    const int j = threadIdx.x & 15;
    const int g = threadIdx.x >> 4;             // 0..15
    const int NPB = NQ / CHUNKS;
    const int nbase = chunk * NPB;

    float op[IC], oacc[IC];
#pragma unroll
    for (int i = 0; i < IC; ++i) {
        op[i]   = o_prev[(b * IC + i) * JC + j];
        oacc[i] = 0.f;
    }

    const __bf16* ub = uhat + (long)b * IC * NQ * JC;

    for (int n = nbase + g; n < nbase + NPB; n += 16) {
        float u[IC], bb[IC];
#pragma unroll
        for (int i = 0; i < IC; ++i)
            u[i] = (float)ub[((long)i * NQ + n) * JC + j];
        // agreement: 16-lane butterfly dot over j
#pragma unroll
        for (int i = 0; i < IC; ++i) {
            float d = op[i] * u[i];
            d += __shfl_xor(d, 1, 16);
            d += __shfl_xor(d, 2, 16);
            d += __shfl_xor(d, 4, 16);
            d += __shfl_xor(d, 8, 16);
            bb[i] = d;
        }
        // softmax over 32 capsules, in-register
        float mx = bb[0];
#pragma unroll
        for (int i = 1; i < IC; ++i) mx = fmaxf(mx, bb[i]);
        float Z = 0.f;
#pragma unroll
        for (int i = 0; i < IC; ++i) { bb[i] = __expf(bb[i] - mx); Z += bb[i]; }
        const float invZ = 1.0f / Z;
#pragma unroll
        for (int i = 0; i < IC; ++i)
            oacc[i] = fmaf(bb[i] * invZ, u[i], oacc[i]);
    }

    // reduce the 16 n-groups through LDS, then one global atomic per (i,j)
    __shared__ float red[IC][JC][16];
#pragma unroll
    for (int i = 0; i < IC; ++i) red[i][j][g] = oacc[i];
    __syncthreads();
    for (int p = threadIdx.x; p < IC * JC; p += 256) {
        const int ii = p >> 4, jj = p & 15;
        float s = 0.f;
#pragma unroll
        for (int gg = 0; gg < 16; ++gg) s += red[ii][jj][gg];
        atomicAdd(&o_next[(b * IC + ii) * JC + jj], s);
    }
}

// ---------------------------------------------------------------------------
// K5: per-(b,i) L2-normalize (mid iterations) or squash (final).
// ---------------------------------------------------------------------------
__global__ void o_finalize(const float* __restrict__ o_raw,
                           float* __restrict__ out, int do_squash) {
    const int bi = blockIdx.x;                  // 1024
    const int j  = threadIdx.x;                 // 16
    float v = o_raw[bi * JC + j];
    float s2 = v * v;
#pragma unroll
    for (int msk = 1; msk < 16; msk <<= 1) s2 += __shfl_xor(s2, msk, 16);
    float scale;
    if (do_squash) { float t = s2 + 1e-7f; scale = __fsqrt_rn(t) / (0.5f + t); }
    else           { scale = rsqrtf(fmaxf(s2, 1e-12f)); }
    out[bi * JC + j] = v * scale;
}

extern "C" void kernel_launch(void* const* d_in, const int* in_sizes, int n_in,
                              void* d_out, int out_size, void* d_ws, size_t ws_size,
                              hipStream_t stream) {
    const float* u_vecs = (const float*)d_in[0];   // [32,4096,256] f32
    const float* W      = (const float*)d_in[1];   // [1,256,512]  f32
    float* out = (float*)d_out;                    // [32,32,16]   f32

    char* ws = (char*)d_ws;
    __bf16* Wp   = (__bf16*)ws;                                    // 256 KB
    __bf16* uhat = (__bf16*)(ws + (1u << 20));                     // 128 MiB
    float*  oA   = (float*)(ws + (1u << 20) + (size_t)BQ*IC*NQ*JC*2);
    float*  oB   = oA + BQ * IC * JC;

    pack_w   <<<512, 256, 0, stream>>>(W, Wp);
    gemm_uhat<<<(BQ * NQ) / 16, 128, 0, stream>>>(u_vecs, Wp, uhat);

    // iter 0: uniform coupling -> mean, then l2-normalize
    o_init<<<BQ * IC, 256, 0, stream>>>(uhat, oA);

    // iter 1
    zero_o    <<<(BQ * IC * JC) / 256, 256, 0, stream>>>(oB);
    route_step<<<dim3(BQ, CHUNKS), 256, 0, stream>>>(uhat, oA, oB);
    o_finalize<<<BQ * IC, JC, 0, stream>>>(oB, oA, 0);

    // iter 2 + squash -> output
    zero_o    <<<(BQ * IC * JC) / 256, 256, 0, stream>>>(oB);
    route_step<<<dim3(BQ, CHUNKS), 256, 0, stream>>>(uhat, oA, oB);
    o_finalize<<<BQ * IC, JC, 0, stream>>>(oB, out, 1);
}
